// LSTMFourierConnectivityDecoder_50130858279030
// MI455X (gfx1250) — matrix operand order repaired and checked by reading, not compiled, not run on hardware
//
#include <hip/hip_runtime.h>
#include <math.h>

#define B_    32
#define T_    512
#define L_    512
#define IN_   256
#define CONN_ 256
#define BT_   (B_ * T_)

typedef __bf16 v16bf __attribute__((ext_vector_type(16)));
typedef float  v8f   __attribute__((ext_vector_type(8)));
typedef int    v4i   __attribute__((ext_vector_type(4)));

struct FragBits { v4i lo, hi; };

__device__ __forceinline__ v16bf ld_frag(const void* p0, const void* p1) {
    FragBits f;
    f.lo = *(const v4i*)p0;
    f.hi = *(const v4i*)p1;
    return __builtin_bit_cast(v16bf, f);
}

__device__ __forceinline__ v8f wmma_bf16(v16bf a, v16bf b, v8f c) {
    // (neg_a, A, neg_b, B, c_mod, C, reuse_a, reuse_b)
    return __builtin_amdgcn_wmma_f32_16x16x32_bf16(false, a, false, b, (short)0, c, false, false);
}

__device__ __forceinline__ unsigned short f2bf(float x) {
    unsigned int u = __builtin_bit_cast(unsigned int, x);
    u += 0x7FFFu + ((u >> 16) & 1u);   // round-to-nearest-even
    return (unsigned short)(u >> 16);
}

__device__ __forceinline__ float sigmoidf_(float x) { return 1.0f / (1.0f + expf(-x)); }

// ---------------- prep kernels ----------------

// x [B, IN, T] f32  ->  xtb [B*T, IN] bf16  (tiled transpose)
__global__ void k_transpose_x(const float* __restrict__ x, unsigned short* __restrict__ xtb) {
    __shared__ float tile[32][33];
    int tx = threadIdx.x, ty = threadIdx.y;
    int t0 = blockIdx.x * 32, i0 = blockIdx.y * 32, b = blockIdx.z;
    const float* xb = x + (size_t)b * IN_ * T_;
    for (int s = 0; s < 4; ++s) {
        int i = i0 + ty + 8 * s;
        tile[ty + 8 * s][tx] = xb[(size_t)i * T_ + t0 + tx];
    }
    __syncthreads();
    for (int s = 0; s < 4; ++s) {
        int t = t0 + ty + 8 * s;
        xtb[((size_t)b * T_ + t) * IN_ + i0 + tx] = f2bf(tile[tx][ty + 8 * s]);
    }
}

// conn [B*T, CONN] f32 -> bf16 (same layout)
__global__ void k_conv_conn(const float* __restrict__ src, unsigned short* __restrict__ dst) {
    size_t id = (size_t)blockIdx.x * 256 + threadIdx.x;
    dst[id] = f2bf(src[id]);
}

// pack row-major W[K,N] f32 into per-lane WMMA B-fragment layout:
// dword index = ((nt*KT + kt)*32 + lane)*8 + v ; lane: n = nt*16 + (lane&15),
// k = kt*32 + (lane>>4)*8 + (v<4 ? 2v : 16+2(v-4)) (+0/+1 packed in dword)
__global__ void k_pack_b(const float* __restrict__ src, int K, int N, int* __restrict__ dst) {
    int id = blockIdx.x * 256 + threadIdx.x;
    int total = (K / 32) * (N / 16) * 256;
    if (id >= total) return;
    int v = id & 7, lane = (id >> 3) & 31, idx2 = id >> 8;
    int KT = K / 32;
    int kt = idx2 % KT, nt = idx2 / KT;
    int k = kt * 32 + ((lane >> 4) << 3) + ((v < 4) ? 2 * v : 16 + 2 * (v - 4));
    int n = nt * 16 + (lane & 15);
    unsigned int lo = f2bf(src[(size_t)k * N + n]);
    unsigned int hi = f2bf(src[(size_t)(k + 1) * N + n]);
    dst[id] = (int)(lo | (hi << 16));
}

// pack the fused gate weight matrix Wbig[1024,1024]:
// rows 0..511 = [Wi | Wg], rows 512..1023 = [Ri | Rg]; cols 0..511 -> i gate, 512..1023 -> f gate
__global__ void k_pack_big(const float* __restrict__ Wi, const float* __restrict__ Ri,
                           const float* __restrict__ Wg, const float* __restrict__ Rg,
                           int* __restrict__ dst) {
    int id = blockIdx.x * 256 + threadIdx.x;           // total = 32*64*256
    int v = id & 7, lane = (id >> 3) & 31, idx2 = id >> 8;
    int kt = idx2 & 31, nt = idx2 >> 5;                // KT = 32
    int k = kt * 32 + ((lane >> 4) << 3) + ((v < 4) ? 2 * v : 16 + 2 * (v - 4));      // always even
    int n = nt * 16 + (lane & 15);
    const float* s = (k < 512) ? ((n < 512) ? Wi : Wg) : ((n < 512) ? Ri : Rg);
    int kk = k & 511, nn = n & 511;
    unsigned int lo = f2bf(s[(size_t)kk * 512 + nn]);
    unsigned int hi = f2bf(s[(size_t)(kk + 1) * 512 + nn]);
    dst[id] = (int)(lo | (hi << 16));
}

// zbuf [B, T+1, L] bf16: row t=0 of each batch <- init_h
__global__ void k_zinit(const float* __restrict__ init_h, unsigned short* __restrict__ zbuf) {
    int id = blockIdx.x * 256 + threadIdx.x;           // B*L threads
    int b = id >> 9, l = id & 511;
    zbuf[(size_t)b * (T_ + 1) * L_ + l] = f2bf(init_h[l]);
}

// ---------------- fused fourier GEMM + LayerNorm + sin ----------------
// block = 32 rows x 512 cols; 8 waves, each 2 M-tiles x 4 N-tiles, K double-buffered
__global__ void k_fourier(const unsigned short* __restrict__ xtb, const int* __restrict__ wfp,
                          const float* __restrict__ bfv, const float* __restrict__ gamma,
                          const float* __restrict__ beta, float* __restrict__ obuf,
                          unsigned short* __restrict__ zbuf) {
    __shared__ float pb[32 * 520];
    __shared__ float smu[32], srs[32];
    int tid = threadIdx.x;
    int w = tid >> 5, ln = tid & 31;
    int half = ln >> 4, m = ln & 15, nq = ln & 15;
    int rbase = blockIdx.x * 32;
    const unsigned short* arow0 = xtb + (size_t)(rbase + m) * IN_;
    const unsigned short* arow1 = arow0 + (size_t)16 * IN_;

    v8f acc[2][4] = {};
    v16bf aB[2][2], bB[2][4];
    {   // preload kt = 0
        int ko = half * 8;
        aB[0][0] = ld_frag(arow0 + ko, arow0 + ko + 16);
        aB[0][1] = ld_frag(arow1 + ko, arow1 + ko + 16);
        for (int j = 0; j < 4; ++j) {
            const int* bp = wfp + (((size_t)(w * 4 + j) * (IN_ / 32)) << 8) + ln * 8;
            bB[0][j] = ld_frag(bp, bp + 4);
        }
    }
#pragma unroll
    for (int kt = 0; kt < IN_ / 32; ++kt) {
        int cur = kt & 1, nxt = cur ^ 1;
        if (kt < IN_ / 32 - 1) {
            int ko = (kt + 1) * 32 + half * 8;
            aB[nxt][0] = ld_frag(arow0 + ko, arow0 + ko + 16);
            aB[nxt][1] = ld_frag(arow1 + ko, arow1 + ko + 16);
            for (int j = 0; j < 4; ++j) {
                const int* bp = wfp + (((size_t)(w * 4 + j) * (IN_ / 32) + kt + 1) << 8) + ln * 8;
                bB[nxt][j] = ld_frag(bp, bp + 4);
            }
        }
        for (int mi = 0; mi < 2; ++mi)
            for (int j = 0; j < 4; ++j)
                acc[mi][j] = wmma_bf16(aB[cur][mi], bB[cur][j], acc[mi][j]);
    }
    // p = cos(acc + bf) into LDS strip
    for (int mi = 0; mi < 2; ++mi)
        for (int j = 0; j < 4; ++j) {
            int n = (w * 4 + j) * 16 + nq;
            float bb = bfv[n];
            for (int v = 0; v < 8; ++v)
                pb[(mi * 16 + v + half * 8) * 520 + n] = cosf(acc[mi][j][v] + bb);
        }
    __syncthreads();
    // row-wise mean/var: wave w handles rows 4w..4w+3
    for (int rr = 0; rr < 2; ++rr) {
        int row = 4 * w + 2 * rr + half;
        float s = 0.f, s2 = 0.f;
        for (int c = nq; c < 512; c += 16) {
            float p = pb[row * 520 + c];
            s += p; s2 += p * p;
        }
        for (int off = 8; off >= 1; off >>= 1) {
            s  += __shfl_xor(s,  off, 32);
            s2 += __shfl_xor(s2, off, 32);
        }
        if (nq == 0) {
            float mu = s * (1.0f / 512.0f);
            float var = s2 * (1.0f / 512.0f) - mu * mu;
            smu[row] = mu;
            srs[row] = rsqrtf(var + 1e-3f);
        }
    }
    __syncthreads();
    // z = gamma*(p-mu)*rstd + beta ; o = sin(z)
    for (int e = 0; e < 64; ++e) {
        int idx = e * 256 + tid;
        int row = idx >> 9, col = idx & 511;
        float p = pb[row * 520 + col];
        float z = gamma[col] * (p - smu[row]) * srs[row] + beta[col];
        int rg = rbase + row;
        obuf[(size_t)rg * L_ + col] = sinf(z);
        int b = rg >> 9, t = rg & 511;
        zbuf[((size_t)b * (T_ + 1) + t + 1) * L_ + col] = f2bf(z);
    }
}

// ---------------- u = conn @ Wc + bc ----------------
// block = 32 rows x 512 cols; 8 waves, each 2 M-tiles x 4 N-tiles, K double-buffered
__global__ void k_conn_gemm(const unsigned short* __restrict__ connb, const int* __restrict__ wcp,
                            const float* __restrict__ bc, float* __restrict__ ubuf) {
    int tid = threadIdx.x;
    int w = tid >> 5, ln = tid & 31;
    int half = ln >> 4, m = ln & 15, nq = ln & 15;
    int rbase = blockIdx.x * 32;
    const unsigned short* arow0 = connb + (size_t)(rbase + m) * CONN_;
    const unsigned short* arow1 = arow0 + (size_t)16 * CONN_;

    v8f acc[2][4] = {};
    v16bf aB[2][2], bB[2][4];
    {
        int ko = half * 8;
        aB[0][0] = ld_frag(arow0 + ko, arow0 + ko + 16);
        aB[0][1] = ld_frag(arow1 + ko, arow1 + ko + 16);
        for (int j = 0; j < 4; ++j) {
            const int* bp = wcp + (((size_t)(w * 4 + j) * (CONN_ / 32)) << 8) + ln * 8;
            bB[0][j] = ld_frag(bp, bp + 4);
        }
    }
#pragma unroll
    for (int kt = 0; kt < CONN_ / 32; ++kt) {
        int cur = kt & 1, nxt = cur ^ 1;
        if (kt < CONN_ / 32 - 1) {
            int ko = (kt + 1) * 32 + half * 8;
            aB[nxt][0] = ld_frag(arow0 + ko, arow0 + ko + 16);
            aB[nxt][1] = ld_frag(arow1 + ko, arow1 + ko + 16);
            for (int j = 0; j < 4; ++j) {
                const int* bp = wcp + (((size_t)(w * 4 + j) * (CONN_ / 32) + kt + 1) << 8) + ln * 8;
                bB[nxt][j] = ld_frag(bp, bp + 4);
            }
        }
        for (int mi = 0; mi < 2; ++mi)
            for (int j = 0; j < 4; ++j)
                acc[mi][j] = wmma_bf16(aB[cur][mi], bB[cur][j], acc[mi][j]);
    }
    for (int mi = 0; mi < 2; ++mi)
        for (int j = 0; j < 4; ++j) {
            int n = (w * 4 + j) * 16 + nq;
            float bb = bc[n];
            for (int v = 0; v < 8; ++v) {
                int rg = rbase + mi * 16 + v + half * 8;
                ubuf[(size_t)rg * L_ + n] = acc[mi][j][v] + bb;
            }
        }
}

// ---------------- fused gate GEMM: [z|z_prev] @ [[Wi Wg],[Ri Rg]] ----------------
// grid (512, 2): blockIdx.x = 32-row strip, blockIdx.y = 512-col half of N=1024
// 8 waves, each 2 M-tiles x 4 N-tiles, K=1024 double-buffered
// out G[B*T,1024]: cols 0..511 = i*u, cols 512..1023 = f
__global__ void k_gates(const unsigned short* __restrict__ zbuf, const int* __restrict__ wbp,
                        const float* __restrict__ bi, const float* __restrict__ bg,
                        const float* __restrict__ ubuf, float* __restrict__ G) {
    int tid = threadIdx.x;
    int w = tid >> 5, ln = tid & 31;
    int half = ln >> 4, m = ln & 15, nq = ln & 15;
    int rbase = blockIdx.x * 32;
    int ntbase = blockIdx.y * 32 + w * 4;

    const unsigned short* zz[2];
    const unsigned short* zp[2];
    for (int mi = 0; mi < 2; ++mi) {
        int r = rbase + mi * 16 + m;
        int b = r >> 9, t = r & 511;
        zz[mi] = zbuf + ((size_t)b * (T_ + 1) + t + 1) * L_;   // z_t
        zp[mi] = zbuf + ((size_t)b * (T_ + 1) + t) * L_;       // z_{t-1}
    }

    v8f acc[2][4] = {};
    v16bf aB[2][2], bB[2][4];
    {
        int ko = half * 8;
        aB[0][0] = ld_frag(zz[0] + ko, zz[0] + ko + 16);
        aB[0][1] = ld_frag(zz[1] + ko, zz[1] + ko + 16);
        for (int j = 0; j < 4; ++j) {
            const int* bp = wbp + (((size_t)(ntbase + j) * 32) << 8) + ln * 8;
            bB[0][j] = ld_frag(bp, bp + 4);
        }
    }
#pragma unroll
    for (int kt = 0; kt < 32; ++kt) {
        int cur = kt & 1, nxt = cur ^ 1;
        if (kt < 31) {
            int kk = (kt + 1) * 32;
            int ko = (kk & 511) + half * 8;
            const unsigned short* a0 = (kk < 512) ? zz[0] : zp[0];
            const unsigned short* a1 = (kk < 512) ? zz[1] : zp[1];
            aB[nxt][0] = ld_frag(a0 + ko, a0 + ko + 16);
            aB[nxt][1] = ld_frag(a1 + ko, a1 + ko + 16);
            for (int j = 0; j < 4; ++j) {
                const int* bp = wbp + (((size_t)(ntbase + j) * 32 + kt + 1) << 8) + ln * 8;
                bB[nxt][j] = ld_frag(bp, bp + 4);
            }
        }
        for (int mi = 0; mi < 2; ++mi)
            for (int j = 0; j < 4; ++j)
                acc[mi][j] = wmma_bf16(aB[cur][mi], bB[cur][j], acc[mi][j]);
    }
    for (int mi = 0; mi < 2; ++mi)
        for (int j = 0; j < 4; ++j) {
            int n = (ntbase + j) * 16 + nq;                // 0..1023
            float bias = (n < 512) ? bi[n] : bg[n - 512];
            for (int v = 0; v < 8; ++v) {
                int rg = rbase + mi * 16 + v + half * 8;
                float g = sigmoidf_(acc[mi][j][v] + bias);
                float outv = (n < 512) ? g * ubuf[(size_t)rg * L_ + n] : g;
                G[(size_t)rg * 1024 + n] = outv;
            }
        }
}

// ---------------- sequential scan + output transpose ----------------
__global__ void k_scan(const float* __restrict__ G, const float* __restrict__ obuf,
                       const float* __restrict__ init_c, float* __restrict__ out) {
    __shared__ float hbuf[256][17];
    int tid = threadIdx.x;
    int b = blockIdx.x >> 1;
    int lseg = blockIdx.x & 1;
    int l = lseg * 256 + tid;
    float c = init_c[l];
    for (int t0 = 0; t0 < T_; t0 += 16) {
        for (int tl = 0; tl < 16; ++tl) {
            size_t ro = (size_t)b * T_ + t0 + tl;
            float f  = G[ro * 1024 + 512 + l];
            float iu = G[ro * 1024 + l];
            c = f * c + iu;
            hbuf[tid][tl] = obuf[ro * L_ + l] * c;
        }
        __syncthreads();
        int tloc = tid & 15, rg = tid >> 4;
        for (int it = 0; it < 16; ++it) {
            int lloc = it * 16 + rg;
            out[((size_t)b * L_ + lseg * 256 + lloc) * T_ + t0 + tloc] = hbuf[lloc][tloc];
        }
        __syncthreads();
    }
}

// ---------------- launch ----------------
extern "C" void kernel_launch(void* const* d_in, const int* in_sizes, int n_in,
                              void* d_out, int out_size, void* d_ws, size_t ws_size,
                              hipStream_t stream) {
    (void)in_sizes; (void)n_in; (void)out_size; (void)ws_size;
    const float* x      = (const float*)d_in[0];
    const float* conn   = (const float*)d_in[1];
    const float* Wf     = (const float*)d_in[2];
    const float* bfv    = (const float*)d_in[3];
    const float* gamma  = (const float*)d_in[4];
    const float* beta   = (const float*)d_in[5];
    const float* Wi     = (const float*)d_in[6];
    const float* Ri     = (const float*)d_in[7];
    const float* bi     = (const float*)d_in[8];
    const float* Wg     = (const float*)d_in[9];
    const float* Rg     = (const float*)d_in[10];
    const float* bg     = (const float*)d_in[11];
    const float* Wc     = (const float*)d_in[12];
    const float* bc     = (const float*)d_in[13];
    const float* init_h = (const float*)d_in[14];
    const float* init_c = (const float*)d_in[15];
    float* out = (float*)d_out;

    char* base = (char*)d_ws;
    size_t off = 0;
    auto carve = [&](size_t bytes) {
        char* p = base + off;
        off = (off + bytes + 255) & ~(size_t)255;
        return p;
    };
    unsigned short* xtb   = (unsigned short*)carve((size_t)BT_ * IN_ * 2);
    unsigned short* connb = (unsigned short*)carve((size_t)BT_ * CONN_ * 2);
    int* wfp = (int*)carve((size_t)(IN_ / 32) * (L_ / 16) * 256 * 4);
    int* wcp = (int*)carve((size_t)(CONN_ / 32) * (L_ / 16) * 256 * 4);
    int* wbp = (int*)carve((size_t)32 * 64 * 256 * 4);
    unsigned short* zbuf = (unsigned short*)carve((size_t)B_ * (T_ + 1) * L_ * 2);
    float* obuf = (float*)carve((size_t)BT_ * L_ * 4);
    float* ubuf = (float*)carve((size_t)BT_ * L_ * 4);
    float* G    = (float*)carve((size_t)BT_ * 1024 * 4);

    k_transpose_x<<<dim3(T_ / 32, IN_ / 32, B_), dim3(32, 8), 0, stream>>>(x, xtb);
    k_conv_conn<<<(BT_ * CONN_) / 256, 256, 0, stream>>>(conn, connb);
    k_pack_b<<<((IN_ / 32) * (L_ / 16) * 256) / 256, 256, 0, stream>>>(Wf, IN_, L_, wfp);
    k_pack_b<<<((CONN_ / 32) * (L_ / 16) * 256) / 256, 256, 0, stream>>>(Wc, CONN_, L_, wcp);
    k_pack_big<<<(32 * 64 * 256) / 256, 256, 0, stream>>>(Wi, Ri, Wg, Rg, wbp);
    k_zinit<<<(B_ * L_) / 256, 256, 0, stream>>>(init_h, zbuf);
    k_fourier<<<BT_ / 32, 256, 0, stream>>>(xtb, wfp, bfv, gamma, beta, obuf, zbuf);
    k_conn_gemm<<<BT_ / 32, 256, 0, stream>>>(connb, wcp, bc, ubuf);
    k_gates<<<dim3(512, 2), 256, 0, stream>>>(zbuf, wbp, bi, bg, ubuf, G);
    k_scan<<<(B_ * L_) / 256, 256, 0, stream>>>(G, obuf, init_c, out);
}
